// Model_39676907887590
// MI455X (gfx1250) — matrix operand order repaired and checked by reading, compiled, not run on hardware
//
#include <hip/hip_runtime.h>
#include <hip/hip_bf16.h>

// Problem dims (fixed by reference): B=4, H=16, S=1024, D=64
#define B_ 4
#define H_ 16
#define S_ 1024
#define D_ 64

typedef __attribute__((ext_vector_type(16))) __bf16 v16bf;
typedef __attribute__((ext_vector_type(8)))  float  v8f;

union FragU {
    v16bf v;
    unsigned int u[8];
};

static __device__ __forceinline__ unsigned short f2bf(float f) {
    unsigned int x = __float_as_uint(f);
    x += 0x7FFFu + ((x >> 16) & 1u);   // round-to-nearest-even
    return (unsigned short)(x >> 16);
}
static __device__ __forceinline__ unsigned int pack2(float lo, float hi) {
    return (unsigned int)f2bf(lo) | ((unsigned int)f2bf(hi) << 16);
}

// ---------------------------------------------------------------------------
// Kernel 1: q3[h][b][sr][sc] = mask!=0 ? mask : (Q·K)/32
// Block: 256 threads (8 waves), computes a 64(M) x 128(N) score tile.
// ---------------------------------------------------------------------------
__global__ __launch_bounds__(256)
void qk_mask_kernel(const float* __restrict__ Q,     // [B][H][S][D]
                    const float* __restrict__ K,     // [H][B][D][S]
                    const float* __restrict__ mask,  // [B][H][S][S]
                    float* __restrict__ q3) {        // [H][B][S][S]
    // bf16 pairs packed in u32: [row][k/2]
    __shared__ unsigned int lq32[64 * 32];    // Q tile  (m, k)   8 KB
    __shared__ unsigned int lk32[128 * 32];   // K tile  (n, k)  16 KB

    const int hb  = blockIdx.z;          // h*B + b  (layout of input2/q3)
    const int h   = hb >> 2;
    const int b   = hb & 3;
    const int bh  = b * H_ + h;          // layout of input/mask
    const int sr0 = blockIdx.y * 64;
    const int sc0 = blockIdx.x * 128;
    const int t   = threadIdx.x;

    // ---- stage Q tile: global f32 [m][d] -> bf16 pairs [m][d/2]
    {
        const float* qsrc = Q + ((long)bh * S_ + sr0) * D_;
        #pragma unroll
        for (int i = 0; i < 8; ++i) {
            int idx = t + i * 256;            // 2048 u32
            int row = idx >> 5;
            int dp  = idx & 31;
            lq32[idx] = pack2(qsrc[row * D_ + 2 * dp],
                              qsrc[row * D_ + 2 * dp + 1]);
        }
    }
    // ---- stage K tile: global f32 [d][s] -> bf16 pairs [s_local][d/2] (transpose)
    {
        const float* ksrc = K + (long)hb * D_ * S_ + sc0;
        #pragma unroll
        for (int i = 0; i < 16; ++i) {
            int p  = t + i * 256;             // 4096 u32
            int dp = p >> 7;                  // 0..31
            int sc = p & 127;                 // coalesced along s
            lk32[sc * 32 + dp] = pack2(ksrc[(long)(2 * dp) * S_ + sc],
                                       ksrc[(long)(2 * dp + 1) * S_ + sc]);
        }
    }
    __syncthreads();

    const int lane = t & 31;
    const int wave = t >> 5;
    const int half = lane >> 4;    // K-group select
    const int ml   = lane & 15;
    const int n0   = wave * 16;    // each wave owns one 16-col strip

    v8f zero = {};
    v8f acc[4];
    acc[0] = zero; acc[1] = zero; acc[2] = zero; acc[3] = zero;

    #pragma unroll
    for (int kc = 0; kc < 2; ++kc) {          // D=64 -> two K=32 chunks
        // B fragment (32x16): lanes 0-15 K=0-7/16-23, lanes 16-31 K=8-15/24-31
        FragU bf;
        {
            int bbase = (n0 + ml) * 32 + kc * 16 + half * 4;
            #pragma unroll
            for (int v = 0; v < 8; ++v) bf.u[v] = lk32[bbase + v + (v >> 2) * 4];
        }
        #pragma unroll
        for (int mt = 0; mt < 4; ++mt) {
            // A fragment (16x32): lanes 0-15 K=0-7/16-23, lanes 16-31 K=8-15/24-31
            FragU a;
            int abase = (mt * 16 + ml) * 32 + kc * 16 + half * 4;
            #pragma unroll
            for (int v = 0; v < 4; ++v) {
                a.u[v]     = lq32[abase + v];
                a.u[v + 4] = lq32[abase + 8 + v];
            }
            acc[mt] = __builtin_amdgcn_wmma_f32_16x16x32_bf16(
                false, a.v, false, bf.v, (short)0, acc[mt], false, false);
        }
    }

    // ---- scale (1/sqrt(S)=1/32), mask-replace, store q3
    const float scale = 0.03125f;
    const int col = sc0 + n0 + ml;
    #pragma unroll
    for (int mt = 0; mt < 4; ++mt) {
        int rb = sr0 + mt * 16 + half * 8;    // C/D layout: VGPR r -> rows r, r+8
        #pragma unroll
        for (int r = 0; r < 8; ++r) {
            int row = rb + r;
            float mv = mask[((long)bh * S_ + row) * S_ + col];
            float sc = acc[mt][r] * scale;
            q3[((long)hb * S_ + row) * S_ + col] = (mv != 0.0f) ? mv : sc;
        }
    }
}

// ---------------------------------------------------------------------------
// Kernel 2: out[h][b][s][d] = softmax(q3, axis=-1) @ V
// Block: 256 threads (8 waves), 32 rows of one (h,b).
// Phase 1: row stats (wave32 shuffle reductions).
// Phase 2: cooperative double-buffered LDS staging of P (exp computed ONCE)
//          and V (coalesced), fragments via ds_load, bf16 WMMA accumulate.
// ---------------------------------------------------------------------------
__global__ __launch_bounds__(256)
void softmax_av_kernel(const float* __restrict__ q3,  // [H][B][S][S]
                       const float* __restrict__ V,   // [H][B][S][D]
                       float* __restrict__ out) {     // [H][B][S][D]
    __shared__ float s_max[32];
    __shared__ float s_inv[32];
    __shared__ unsigned int pbuf[2][32 * 16];  // P bf16 pairs [row][k/2]  (2x2 KB)
    __shared__ unsigned int vbuf[2][64 * 16];  // V bf16 pairs [n][k/2]    (2x4 KB)

    const int hb   = blockIdx.y;
    const int r0   = blockIdx.x * 32;
    const int t    = threadIdx.x;
    const int lane = t & 31;
    const int wave = t >> 5;

    const float* qbase = q3 + ((long)hb * S_ + r0) * S_;
    const float* vsrc  = V + (long)hb * S_ * D_;

    // ---- Phase 1: per-row max and 1/sum(exp) (each wave owns 4 rows)
    #pragma unroll
    for (int rr = 0; rr < 4; ++rr) {
        int lr = wave * 4 + rr;
        const float* qrow = qbase + (long)lr * S_;
        float mx = -3.0e38f;
        for (int c = lane; c < S_; c += 32) mx = fmaxf(mx, qrow[c]);
        #pragma unroll
        for (int off = 16; off > 0; off >>= 1) mx = fmaxf(mx, __shfl_xor(mx, off, 32));
        float sm = 0.0f;
        for (int c = lane; c < S_; c += 32) sm += __expf(qrow[c] - mx);
        #pragma unroll
        for (int off = 16; off > 0; off >>= 1) sm += __shfl_xor(sm, off, 32);
        if (lane == 0) { s_max[lr] = mx; s_inv[lr] = 1.0f / sm; }
    }
    __syncthreads();

    // ---- Phase 2: P @ V, K = S = 1024 in 32-chunks
    const int half  = lane >> 4;
    const int ml    = lane & 15;
    const int mtile = wave >> 2;            // 0/1  -> 16-row block
    const int n0    = (wave & 3) * 16;      // 16-col strip of D=64

    v8f acc = {};
    for (int kc = 0; kc < S_ / 32; ++kc) {
        const int k0 = kc * 32;
        unsigned int* pb = pbuf[kc & 1];
        unsigned int* vb = vbuf[kc & 1];

        // stage P tile (32 rows x 32 k) as bf16 pairs: exp computed once
        #pragma unroll
        for (int i = 0; i < 2; ++i) {
            int p   = t + i * 256;            // 512 u32
            int row = p >> 4;
            int kp  = p & 15;
            float mx  = s_max[row];
            float inv = s_inv[row];
            const float* qr = qbase + (long)row * S_ + k0 + 2 * kp;
            pb[p] = pack2(__expf(qr[0] - mx) * inv,
                          __expf(qr[1] - mx) * inv);
        }
        // stage V tile (32 k x 64 n) transposed to [n][k/2], coalesced along n
        #pragma unroll
        for (int i = 0; i < 4; ++i) {
            int p  = t + i * 256;             // 1024 u32
            int kp = p >> 6;
            int n  = p & 63;
            const float* vr = vsrc + (long)(k0 + 2 * kp) * D_ + n;
            vb[n * 16 + kp] = pack2(vr[0], vr[D_]);
        }
        __syncthreads();   // also fences reuse of the other buffer (see analysis)

        FragU a, bfr;
        int abase = (mtile * 16 + ml) * 16 + half * 4;
        #pragma unroll
        for (int v = 0; v < 4; ++v) {
            a.u[v]     = pb[abase + v];
            a.u[v + 4] = pb[abase + 8 + v];
        }
        int bbase = (n0 + ml) * 16 + half * 4;
        #pragma unroll
        for (int v = 0; v < 8; ++v) bfr.u[v] = vb[bbase + v + (v >> 2) * 4];

        acc = __builtin_amdgcn_wmma_f32_16x16x32_bf16(
            false, a.v, false, bfr.v, (short)0, acc, false, false);
    }

    const int col = n0 + ml;
    const int rb  = r0 + mtile * 16 + half * 8;
    #pragma unroll
    for (int r = 0; r < 8; ++r) {
        out[((long)hb * S_ + (rb + r)) * D_ + col] = acc[r];
    }
}

// ---------------------------------------------------------------------------
extern "C" void kernel_launch(void* const* d_in, const int* in_sizes, int n_in,
                              void* d_out, int out_size, void* d_ws, size_t ws_size,
                              hipStream_t stream) {
    (void)in_sizes; (void)n_in; (void)out_size; (void)d_ws; (void)ws_size;
    const float* Q    = (const float*)d_in[0];   // [B,H,S,D]
    const float* Km   = (const float*)d_in[1];   // [H,B,D,S]
    const float* Vm   = (const float*)d_in[2];   // [H,B,S,D]
    const float* mask = (const float*)d_in[3];   // [B,H,S,S]

    float* q3   = (float*)d_out;                               // [H,B,S,S]
    float* outp = (float*)d_out + (long)H_ * B_ * S_ * S_;     // [H,B,S,D]

    dim3 g1(S_ / 128, S_ / 64, B_ * H_);
    qk_mask_kernel<<<g1, dim3(256), 0, stream>>>(Q, Km, mask, q3);

    dim3 g2(S_ / 32, H_ * B_);
    softmax_av_kernel<<<g2, dim3(256), 0, stream>>>(q3, Vm, outp);
}
